// Decoder_60902636257930
// MI455X (gfx1250) — matrix operand order repaired and checked by reading, not compiled, not run on hardware
//
#include <hip/hip_runtime.h>
#include <hip/hip_bf16.h>

typedef _Float16 half_t;
typedef __attribute__((ext_vector_type(16))) _Float16 v16h;
typedef __attribute__((ext_vector_type(8)))  _Float16 v8h;
typedef __attribute__((ext_vector_type(8)))  float    v8f;

#define B_    16
#define S_    512
#define NTREE 31
#define D_    256
#define V_    64
#define ROWS  (B_ * S_ * NTREE)   // 253952
#define WAVES 4
#define MW    32                  // rows per wave (two 16-row WMMA sub-tiles)
#define CS    264                 // f16 activation row stride (halves); 528B, 16B-aligned
#define XS    257                 // f32 buffer row stride (floats)
#define KT8   8                   // 256 / 32
#define RND_HALVES 16384          // 32KB round: 4 column tiles x 8 kt x 32 lanes x 16 halves
#define NROUNDS 13                // 4 (W_feats) + 4 (W1) + 4 (W2) + 1 (W_out)

// ---------------- workspace layout (in halves) ----------------
#define WS_EMB   0
#define WS_WF    (WS_EMB + 4096 * 256)          // 1048576
#define WS_W1    (WS_WF  + 16 * KT8 * 32 * 16)  // +65536
#define WS_W2    (WS_W1  + 16 * KT8 * 32 * 16)
#define WS_WO    (WS_W2  + 16 * KT8 * 32 * 16)

__device__ __forceinline__ float gelu_exact(float x) {
    return 0.5f * x * (1.0f + erff(x * 0.70710678118654752f));
}

// ---------------- prepack kernels ----------------
__global__ void pack_emb_kernel(const float* __restrict__ emb, half_t* __restrict__ dst, int total) {
    int i = blockIdx.x * blockDim.x + threadIdx.x;
    if (i < total) dst[i] = (half_t)emb[i];
}

// Pack row-major f32 W[K][N] into WMMA-B fragment order:
// out[((nt*KT + kt)*32 + lane)*16 + e];
//   col = nt*16 + (lane&15); k = kt*32 + (lane>>4)*8 + (e<8 ? e : e+8)
__global__ void pack_wmma_b_kernel(const float* __restrict__ W, half_t* __restrict__ dst,
                                   int K, int N) {
    int i = blockIdx.x * blockDim.x + threadIdx.x;
    int KT = K / 32;
    int total = (N / 16) * KT * 32 * 16;
    if (i >= total) return;
    int e    = i & 15;
    int lane = (i >> 4) & 31;
    int kt   = (i >> 9) % KT;
    int nt   = (i >> 9) / KT;
    int col  = nt * 16 + (lane & 15);
    int k    = kt * 32 + ((lane >> 4) * 8) + (e < 8 ? e : e + 8);
    dst[i] = (half_t)W[k * N + col];
}

// ---------------- fused decoder helpers ----------------
__device__ __forceinline__ void load_a_frags(const half_t* __restrict__ curSub, v16h a8[KT8], int lane) {
    const int row = lane & 15;
    const int kb  = (lane >> 4) * 8;
#pragma unroll
    for (int kt = 0; kt < KT8; ++kt) {
        const half_t* p = curSub + row * CS + kt * 32 + kb;
        union { v16h v; v8h h[2]; } u;
        u.h[0] = *(const v8h*)(p);
        u.h[1] = *(const v8h*)(p + 16);
        a8[kt] = u.v;
    }
}

__device__ __forceinline__ v16h load_b_frag(const half_t* __restrict__ base, int kt) {
    const half_t* p = base + kt * (32 * 16);   // 1024B stride between kt fragments
    union { v16h v; v8h h[2]; } u;
    u.h[0] = *(const v8h*)(p);
    u.h[1] = *(const v8h*)(p + 8);
    return u.v;
}

// Two 16-row sub-tiles share each B fragment; one-deep pipeline on the LDS B loads.
__device__ __forceinline__ void gemm_tile2(const v16h a0[KT8], const v16h a1[KT8],
                                           const half_t* __restrict__ wB, int ntl, int lane,
                                           v8f& acc0, v8f& acc1) {
    const half_t* base = wB + (ntl * KT8 * 32 + lane) * 16;
    v16h b = load_b_frag(base, 0);
#pragma unroll
    for (int kt = 0; kt < KT8; ++kt) {
        v16h bn = b;
        if (kt < KT8 - 1) bn = load_b_frag(base, kt + 1);
        acc0 = __builtin_amdgcn_wmma_f32_16x16x32_f16(false, a0[kt], false, b, (short)0, acc0, false, false);
        acc1 = __builtin_amdgcn_wmma_f32_16x16x32_f16(false, a1[kt], false, b, (short)0, acc1, false, false);
        b = bn;
    }
}

// One lane per row: full-row LN, f16 result into cur.
__device__ __forceinline__ void layernorm_to_cur(const float* __restrict__ X, half_t* __restrict__ cur,
                                                 const float* __restrict__ g, const float* __restrict__ b,
                                                 int lane) {
    const float* xr = X + lane * XS;
    float s = 0.f, s2 = 0.f;
    for (int k = 0; k < D_; ++k) { float v = xr[k]; s += v; s2 += v * v; }
    const float mu  = s * (1.0f / 256.0f);
    const float var = s2 * (1.0f / 256.0f) - mu * mu;
    const float rs  = rsqrtf(var + 1e-5f);
    half_t* cr = cur + lane * CS;
    for (int k = 0; k < D_; ++k)
        cr[k] = (half_t)((xr[k] - mu) * rs * g[k] + b[k]);
}

__global__ __launch_bounds__(WAVES * 32) void fused_decoder_kernel(
    const half_t* __restrict__ embh,
    const half_t* __restrict__ wfp, const half_t* __restrict__ w1p,
    const half_t* __restrict__ w2p, const half_t* __restrict__ wop,
    const int*    __restrict__ feat_idx,
    const float*  __restrict__ memory,
    const float*  __restrict__ bfeats, const float* __restrict__ b1,
    const float*  __restrict__ b2,
    const float*  __restrict__ lng, const float* __restrict__ lnb,
    float* __restrict__ out) {
    extern __shared__ __align__(16) char smem[];
    const int waveId = threadIdx.x >> 5;
    const int lane   = threadIdx.x & 31;

    // LDS map: [buf0 32KB][buf1 32KB][cur 4x16896][X 4x32896] = 264704 B
    half_t* cur = (half_t*)smem + 2 * RND_HALVES + waveId * (MW * CS);
    float*  X   = (float*)(smem + (2 * RND_HALVES + WAVES * MW * CS) * (int)sizeof(half_t))
                  + waveId * (MW * XS);
    const unsigned ldsBase = (unsigned)(uintptr_t)smem;   // low 32 bits of flat = LDS byte offset
    const int waveBase = (blockIdx.x * WAVES + waveId) * MW;

    const int colL = lane & 15;
    const int mb   = (lane >> 4) * 8;

    // Async global->LDS prefetch of one 32KB weight round into buffer `par`.
    // Each lane moves 16B per issue; 16 issues x 128 lanes = 32KB.
    auto prefetch = [&](int g, int par) {
        const half_t* s;
        if (g < 4)       s = wfp + g * RND_HALVES;
        else if (g < 8)  s = w1p + (g - 4) * RND_HALVES;
        else if (g < 12) s = w2p + (g - 8) * RND_HALVES;
        else             s = wop;
        const unsigned dbase = ldsBase + (unsigned)(par * (RND_HALVES * 2)) + threadIdx.x * 16u;
        const unsigned vbase = threadIdx.x * 16u;
#pragma unroll
        for (int i = 0; i < 16; ++i) {
            asm volatile("global_load_async_to_lds_b128 %0, %1, %2"
                         :: "v"(dbase + (unsigned)(i * 2048)),
                            "v"(vbase + (unsigned)(i * 2048)),
                            "s"(s)
                         : "memory");
        }
    };

    // ---- Stage 0a: gather 32 embedding rows (f16) into cur; prefetch round 0 ----
    prefetch(0, 0);
    for (int c = lane; c < MW * 32; c += 32) {
        const int row   = c >> 5;
        const int chunk = c & 31;
        const int idx   = feat_idx[waveBase + row];
        *(uint4*)(cur + row * CS + chunk * 8) =
            *(const uint4*)(embh + (long)idx * 256 + chunk * 8);
    }

    v16h a0[KT8], a1[KT8];
    int g = 0, parity = 0;

    // ================= Stage 0: x = gelu(A0 @ W_feats + b) + memory =================
    load_a_frags(cur, a0, lane);
    load_a_frags(cur + 16 * CS, a1, lane);
#pragma unroll 1
    for (int r = 0; r < 4; ++r) {
        asm volatile("s_wait_asynccnt 0" ::: "memory");
        __syncthreads();
        if (g + 1 < NROUNDS) prefetch(g + 1, parity ^ 1);
        const half_t* wB = (half_t*)smem + parity * RND_HALVES;
#pragma unroll 1
        for (int ntl = 0; ntl < 4; ++ntl) {
            v8f acc0 = {}, acc1 = {};
            gemm_tile2(a0, a1, wB, ntl, lane, acc0, acc1);
            const int col = (r * 4 + ntl) * 16 + colL;
            const float bias = bfeats[col];
#pragma unroll
            for (int rr = 0; rr < 8; ++rr) {
                const int m0 = mb + rr, m1 = 16 + mb + rr;
                X[m0 * XS + col] = gelu_exact(acc0[rr] + bias)
                                   + memory[((waveBase + m0) / NTREE) * D_ + col];
                X[m1 * XS + col] = gelu_exact(acc1[rr] + bias)
                                   + memory[((waveBase + m1) / NTREE) * D_ + col];
            }
        }
        parity ^= 1; ++g;
    }
    layernorm_to_cur(X, cur, lng, lnb, lane);

    // ================= Stage 1: h = gelu(LN(x) @ W1 + b1) =================
    load_a_frags(cur, a0, lane);
    load_a_frags(cur + 16 * CS, a1, lane);
#pragma unroll 1
    for (int r = 0; r < 4; ++r) {
        asm volatile("s_wait_asynccnt 0" ::: "memory");
        __syncthreads();
        if (g + 1 < NROUNDS) prefetch(g + 1, parity ^ 1);
        const half_t* wB = (half_t*)smem + parity * RND_HALVES;
#pragma unroll 1
        for (int ntl = 0; ntl < 4; ++ntl) {
            v8f acc0 = {}, acc1 = {};
            gemm_tile2(a0, a1, wB, ntl, lane, acc0, acc1);
            const int col = (r * 4 + ntl) * 16 + colL;
            const float bias = b1[col];
#pragma unroll
            for (int rr = 0; rr < 8; ++rr) {
                X[(mb + rr) * XS + col]      = gelu_exact(acc0[rr] + bias);
                X[(16 + mb + rr) * XS + col] = gelu_exact(acc1[rr] + bias);
            }
        }
        parity ^= 1; ++g;
    }
    layernorm_to_cur(X, cur, lng, lnb, lane);

    // ================= Stage 2: h2 = gelu(LN(h) @ W2 + b2) -> cur(f16) =================
    load_a_frags(cur, a0, lane);
    load_a_frags(cur + 16 * CS, a1, lane);
#pragma unroll 1
    for (int r = 0; r < 4; ++r) {
        asm volatile("s_wait_asynccnt 0" ::: "memory");
        __syncthreads();
        if (g + 1 < NROUNDS) prefetch(g + 1, parity ^ 1);
        const half_t* wB = (half_t*)smem + parity * RND_HALVES;
#pragma unroll 1
        for (int ntl = 0; ntl < 4; ++ntl) {
            v8f acc0 = {}, acc1 = {};
            gemm_tile2(a0, a1, wB, ntl, lane, acc0, acc1);
            const int col = (r * 4 + ntl) * 16 + colL;
            const float bias = b2[col];
#pragma unroll
            for (int rr = 0; rr < 8; ++rr) {
                cur[(mb + rr) * CS + col]      = (half_t)gelu_exact(acc0[rr] + bias);
                cur[(16 + mb + rr) * CS + col] = (half_t)gelu_exact(acc1[rr] + bias);
            }
        }
        parity ^= 1; ++g;
    }

    // ================= Head: logits = h2 @ W_out (one 32KB round, 4 tiles) =================
    load_a_frags(cur, a0, lane);
    load_a_frags(cur + 16 * CS, a1, lane);
    asm volatile("s_wait_asynccnt 0" ::: "memory");
    __syncthreads();
    {
        const half_t* wB = (half_t*)smem + parity * RND_HALVES;
#pragma unroll 1
        for (int ntl = 0; ntl < 4; ++ntl) {
            v8f acc0 = {}, acc1 = {};
            gemm_tile2(a0, a1, wB, ntl, lane, acc0, acc1);
            const int col = ntl * 16 + colL;
#pragma unroll
            for (int rr = 0; rr < 8; ++rr) {
                X[(mb + rr) * XS + col]      = acc0[rr];
                X[(16 + mb + rr) * XS + col] = acc1[rr];
            }
        }
    }

    // ---- Softmax over V=64; one lane per row (wave-local X) ----
    {
        float* lr = X + lane * XS;
        float mx = -3.402823466e38f;
        for (int j = 0; j < V_; ++j) mx = fmaxf(mx, lr[j]);
        float sum = 0.f;
        for (int j = 0; j < V_; ++j) { float e = expf(lr[j] - mx); lr[j] = e; sum += e; }
        const float inv = 1.0f / sum;
        float* op = out + (long)(waveBase + lane) * V_;
        for (int j = 0; j < V_; ++j) op[j] = lr[j] * inv;
    }
}

extern "C" void kernel_launch(void* const* d_in, const int* in_sizes, int n_in,
                              void* d_out, int out_size, void* d_ws, size_t ws_size,
                              hipStream_t stream) {
    (void)in_sizes; (void)n_in; (void)out_size; (void)ws_size;
    const float* memory   = (const float*)d_in[0];
    const int*   feat_idx = (const int*)  d_in[1];
    const float* emb      = (const float*)d_in[2];
    const float* W_feats  = (const float*)d_in[3];
    const float* b_feats  = (const float*)d_in[4];
    const float* ln_g     = (const float*)d_in[5];
    const float* ln_b     = (const float*)d_in[6];
    const float* W1       = (const float*)d_in[7];
    const float* b1       = (const float*)d_in[8];
    const float* W2       = (const float*)d_in[9];
    const float* b2       = (const float*)d_in[10];
    const float* W_out    = (const float*)d_in[11];
    float* out = (float*)d_out;

    half_t* ws   = (half_t*)d_ws;
    half_t* embh = ws + WS_EMB;
    half_t* wfp  = ws + WS_WF;
    half_t* w1p  = ws + WS_W1;
    half_t* w2p  = ws + WS_W2;
    half_t* wop  = ws + WS_WO;

    {
        int total = 4096 * 256;
        pack_emb_kernel<<<(total + 255) / 256, 256, 0, stream>>>(emb, embh, total);
    }
    {
        int total = 16 * KT8 * 32 * 16; // 65536
        pack_wmma_b_kernel<<<(total + 255) / 256, 256, 0, stream>>>(W_feats, wfp, D_, D_);
        pack_wmma_b_kernel<<<(total + 255) / 256, 256, 0, stream>>>(W1,      w1p, D_, D_);
        pack_wmma_b_kernel<<<(total + 255) / 256, 256, 0, stream>>>(W2,      w2p, D_, D_);
        int totalo = 4 * KT8 * 32 * 16; // 16384
        pack_wmma_b_kernel<<<(totalo + 255) / 256, 256, 0, stream>>>(W_out,  wop, D_, V_);
    }

    // LDS: 2x32KB double-buffered weight stage + 4 waves * (32*264 f16 + 32*257 f32) = 264704 B
    const size_t smem = (size_t)(2 * RND_HALVES) * sizeof(half_t)
                      + (size_t)WAVES * (MW * CS * sizeof(half_t) + MW * XS * sizeof(float));
    (void)hipFuncSetAttribute((const void*)fused_decoder_kernel,
                              hipFuncAttributeMaxDynamicSharedMemorySize, (int)smem);
    const int grid = ROWS / (MW * WAVES);   // 1984, exact
    fused_decoder_kernel<<<grid, WAVES * 32, smem, stream>>>(
        embh, wfp, w1p, w2p, wop, feat_idx, memory,
        b_feats, b1, b2, ln_g, ln_b, out);
}
